// ArcLengthLoss_747324309673
// MI455X (gfx1250) — matrix operand order repaired and checked by reading, not compiled, not run on hardware
//
#include <hip/hip_runtime.h>
#include <math.h>

typedef __attribute__((ext_vector_type(16))) _Float16 v16h;
typedef __attribute__((ext_vector_type(8)))  float    v8f;

#define NPAIRS   28
#define NB       219          // number of B residual rows (subsets of {0..7}, size >= 3)
#define NBPAD    224          // padded to 14 tiles of 16 columns
#define NTILES_N 14
#define ALPHA_C  0.5f

// pair slot index for anchor pair (a,b), a<b, under the reference's ordering:
// gaps 2..7 first (offsets 0,6,11,15,18,20), then adjacent pairs at 21..27.
__device__ __forceinline__ int pair_idx(int a, int b) {
  int gap = b - a;
  if (gap == 1) return 21 + a;
  return (gap - 2) * 8 - (gap * (gap - 1) / 2 - 1) + a;
}

__global__ __launch_bounds__(256) void arclen_main(const float* __restrict__ x,
                                                   int batch, int tiles,
                                                   float* __restrict__ partials) {
  __shared__ int   smMask[NBPAD];
  __shared__ float smRed[256];

  // Enumerate all subset masks with popcount>=3 (order is irrelevant: mean of |.|).
  if (threadIdx.x == 0) {
    int cnt = 0;
    for (int m = 1; m < 256; ++m)
      if (__popc(m) >= 3) smMask[cnt++] = m;
    for (; cnt < NBPAD; ++cnt) smMask[cnt] = 0;
  }
  __syncthreads();

  const int  lane  = threadIdx.x & 31;
  const int  col16 = lane & 15;
  const bool lo    = lane < 16;
  const int  kbase = lo ? 0 : 8;    // 16-bit B 32x16 layout: lanes 0-15 K{0..7,16..23}, lanes 16-31 K{8..15,24..31}

  // Build the 14 B fragments (M^T tiles, entries in {+1,-1,0}, exact in f16).
  // Same for every batch tile -> resident in VGPRs for the whole kernel.
  v16h Bfrag[NTILES_N];
#pragma unroll
  for (int t = 0; t < NTILES_N; ++t) {
    int m = smMask[t * 16 + col16];
    unsigned pos = 0u, neg = 0u;
    if (__popc(m) >= 3) {
      int first = __ffs(m) - 1;
      int last  = 31 - __clz((unsigned)m);
      pos = 1u << pair_idx(first, last);
      int prev = first;
      int mm = m & (m - 1);           // drop lowest set bit
      while (mm) {
        int nxt = __ffs(mm) - 1;
        neg |= 1u << pair_idx(prev, nxt);
        prev = nxt;
        mm &= mm - 1;
      }
    }
    v16h b;
#pragma unroll
    for (int i = 0; i < 16; ++i) {
      int k = kbase + i + ((i >= 8) ? 8 : 0);
      float v = 0.f;
      if (k < NPAIRS) v = ((pos >> k) & 1u) ? 1.f : (((neg >> k) & 1u) ? -1.f : 0.f);
      b[i] = (_Float16)v;
    }
    Bfrag[t] = b;
  }

  const int wid    = (int)((blockIdx.x * blockDim.x + threadIdx.x) >> 5);
  const int nwaves = (int)((gridDim.x * blockDim.x) >> 5);

  float aAcc = 0.f;   // sum of exp(-x) over adjacent-pair columns (cols 21..27)
  float bAcc = 0.f;   // sum of |x @ M^T| over all rows/cols handled by this lane

  for (int tile = wid; tile < tiles; tile += nwaves) {
    int   row  = tile * 16 + col16;
    float mval = (row < batch) ? 1.f : 0.f;
    size_t roff = (size_t)((row < batch) ? row : 0) * NPAIRS;
    const float4* rp = (const float4*)(x + roff);   // rows are 112B -> 16B aligned chunks

    // A-matrix 16x32 f16 layout: lo lanes K{0..7,16..23}, hi lanes K{8..15,24..31}
    float4 p0 = rp[lo ? 0 : 2];
    float4 p1 = rp[lo ? 1 : 3];
    float4 q0 = rp[lo ? 4 : 6];
    float4 q1;
    if (lo) q1 = rp[5];
    else    q1 = make_float4(0.f, 0.f, 0.f, 0.f);  // K=28..31 pad

    // A-term: cols 21..23 sit in lo lanes' q1.yzw; cols 24..27 in hi lanes' q0
    float ae = lo ? (__expf(-q1.y) + __expf(-q1.z) + __expf(-q1.w))
                  : (__expf(-q0.x) + __expf(-q0.y) + __expf(-q0.z) + __expf(-q0.w));
    aAcc += mval * ae;

    v16h a;
    a[0]  = (_Float16)(mval * p0.x);  a[1]  = (_Float16)(mval * p0.y);
    a[2]  = (_Float16)(mval * p0.z);  a[3]  = (_Float16)(mval * p0.w);
    a[4]  = (_Float16)(mval * p1.x);  a[5]  = (_Float16)(mval * p1.y);
    a[6]  = (_Float16)(mval * p1.z);  a[7]  = (_Float16)(mval * p1.w);
    a[8]  = (_Float16)(mval * q0.x);  a[9]  = (_Float16)(mval * q0.y);
    a[10] = (_Float16)(mval * q0.z);  a[11] = (_Float16)(mval * q0.w);
    a[12] = (_Float16)(mval * q1.x);  a[13] = (_Float16)(mval * q1.y);
    a[14] = (_Float16)(mval * q1.z);  a[15] = (_Float16)(mval * q1.w);

    v8f c = {0.f, 0.f, 0.f, 0.f, 0.f, 0.f, 0.f, 0.f};
#pragma unroll
    for (int t = 0; t < NTILES_N; ++t) {
      v8f d = __builtin_amdgcn_wmma_f32_16x16x32_f16(
          /*neg_a=*/false, a, /*neg_b=*/false, Bfrag[t],
          /*c_mod=*/(short)0, c, /*reuse_a=*/false, /*reuse_b=*/false);
      bAcc += __builtin_fabsf(d[0]) + __builtin_fabsf(d[1]) +
              __builtin_fabsf(d[2]) + __builtin_fabsf(d[3]) +
              __builtin_fabsf(d[4]) + __builtin_fabsf(d[5]) +
              __builtin_fabsf(d[6]) + __builtin_fabsf(d[7]);
    }
  }

  float contrib = aAcc * (ALPHA_C / (7.f * (float)batch)) +
                  bAcc * ((1.f - ALPHA_C) / ((float)NB * (float)batch));

  smRed[threadIdx.x] = contrib;
  __syncthreads();
  for (int w = 128; w > 0; w >>= 1) {
    if ((int)threadIdx.x < w) smRed[threadIdx.x] += smRed[threadIdx.x + w];
    __syncthreads();
  }
  if (threadIdx.x == 0) partials[blockIdx.x] = smRed[0];
}

// Deterministic fixed-order final reduction (no float atomics).
__global__ __launch_bounds__(256) void arclen_reduce(const float* __restrict__ partials,
                                                     int n, float* __restrict__ out) {
  __shared__ float s[256];
  float acc = 0.f;
  for (int i = threadIdx.x; i < n; i += 256) acc += partials[i];
  s[threadIdx.x] = acc;
  __syncthreads();
  for (int w = 128; w > 0; w >>= 1) {
    if ((int)threadIdx.x < w) s[threadIdx.x] += s[threadIdx.x + w];
    __syncthreads();
  }
  if (threadIdx.x == 0) out[0] = s[0];
}

extern "C" void kernel_launch(void* const* d_in, const int* in_sizes, int n_in,
                              void* d_out, int out_size, void* d_ws, size_t ws_size,
                              hipStream_t stream) {
  const float* x = (const float*)d_in[0];
  int batch = in_sizes[0] / NPAIRS;          // 500000
  int tiles = (batch + 15) / 16;             // 31250 row-tiles of 16

  int blocks = (tiles + 7) / 8;              // 8 waves (256 thr) per block, 1 tile/wave
  if (blocks > 3072) blocks = 3072;          // grid-stride beyond this
  int maxPart = (int)(ws_size / sizeof(float));
  if (maxPart >= 1 && blocks > maxPart) blocks = maxPart;
  if (blocks < 1) blocks = 1;

  arclen_main<<<blocks, 256, 0, stream>>>(x, batch, tiles, (float*)d_ws);
  arclen_reduce<<<1, 256, 0, stream>>>((const float*)d_ws, blocks, (float*)d_out);
}
